// SDFNetwork_2d_hash_61203283968104
// MI455X (gfx1250) — matrix operand order repaired and checked by reading, not compiled, not run on hardware
//
#include <hip/hip_runtime.h>
#include <hip/hip_fp16.h>

typedef __attribute__((ext_vector_type(16))) _Float16 v16h;
typedef __attribute__((ext_vector_type(8)))  _Float16 v8h;
typedef __attribute__((ext_vector_type(8)))  float    v8f;

#define NLV 16
#define LOG2_T 19
#define TSIZE (1u << LOG2_T)

// scale_l = 16*1.5^l - 1 ; res_l = ceil(scale)+1 ; dense iff res*res <= 2^19 (res <= 724)
__constant__ float kScale[NLV] = {
    15.0f, 23.0f, 35.0f, 53.0f, 80.0f, 120.5f, 181.25f, 272.375f,
    409.0625f, 614.09375f, 921.640625f, 1382.9609375f,
    2074.94140625f, 3112.912109375f, 4669.8681640625f, 7005.30224609375f };
__constant__ int kRes[NLV] = {
    16, 24, 36, 54, 81, 122, 183, 274, 411, 616, 923, 1384, 2076, 3114, 4671, 7007 };

__device__ inline v8f v8f_zero() {
    v8f v;
#pragma unroll
    for (int i = 0; i < 8; ++i) v[i] = 0.0f;
    return v;
}

__device__ inline v8f wmma16(v16h a, v16h b, v8f c) {
    // v_wmma_f32_16x16x32_f16  D = A*B + C
    return __builtin_amdgcn_wmma_f32_16x16x32_f16(
        /*neg_a=*/false, a, /*neg_b=*/false, b,
        /*c_mod=*/(short)0, c, /*reuse_a=*/false, /*reuse_b=*/false);
}

// A operand: two contiguous 16B chunks (k = off..off+7 and off+16..off+23)
__device__ inline v16h ldA(const _Float16* base, int off /*halves, 8-half aligned*/) {
    v8h lo = *(const v8h*)(base + off);
    v8h hh = *(const v8h*)(base + off + 16);
    return __builtin_shufflevector(lo, hh, 0,1,2,3,4,5,6,7,8,9,10,11,12,13,14,15);
}

// B operand: two contiguous 16B chunks (k = off..off+7 and off+16..off+23)
__device__ inline v16h ldB(const _Float16* base, int off /*halves, 8-half aligned*/) {
    v8h lo = *(const v8h*)(base + off);
    v8h hh = *(const v8h*)(base + off + 16);
    return __builtin_shufflevector(lo, hh, 0,1,2,3,4,5,6,7,8,9,10,11,12,13,14,15);
}

__device__ inline float sigm(float v) {
    return __builtin_amdgcn_rcpf(1.0f + __expf(-v));
}

__global__ __launch_bounds__(256)
void ngp_fused_kernel(const float* __restrict__ inp,      // [N,3]
                      const float* __restrict__ tables,   // [16, 2^19, 2]
                      const float* __restrict__ Wtiny,    // [32,65]
                      const float* __restrict__ btiny,    // [65]
                      const float* __restrict__ Wp1,      // [12,64]
                      const float* __restrict__ Wp2,      // [64,64]
                      const float* __restrict__ Wp3,      // [64,1]
                      float* __restrict__ out,            // [N,65]
                      int Npts, int numTiles)
{
    // ---- LDS: transposed f16 weights + per-wave activation staging tile ----
    __shared__ _Float16 sW1t[64 * 32];   // [n=64][k=32], k>=12 zero
    __shared__ _Float16 sW2t[64 * 64];   // [n=64][k=64]
    __shared__ _Float16 sW3t[16 * 64];   // [n=16][k=64], n>0 zero
    __shared__ _Float16 sWtt[80 * 32];   // [n=80][k=32], n>=65 zero
    __shared__ float    sBt[80];
    __shared__ _Float16 sAct[8][16 * 64];

    const int tid = threadIdx.x;
    for (int i = tid; i < 64 * 32; i += 256) {
        int n = i >> 5, k = i & 31;
        sW1t[i] = (k < 12) ? (_Float16)Wp1[k * 64 + n] : (_Float16)0.0f;
    }
    for (int i = tid; i < 64 * 64; i += 256) {
        int n = i >> 6, k = i & 63;
        sW2t[i] = (_Float16)Wp2[k * 64 + n];
    }
    for (int i = tid; i < 16 * 64; i += 256) {
        int n = i >> 6, k = i & 63;
        sW3t[i] = (n == 0) ? (_Float16)Wp3[k] : (_Float16)0.0f;
    }
    for (int i = tid; i < 80 * 32; i += 256) {
        int n = i >> 5, k = i & 31;
        sWtt[i] = (n < 65) ? (_Float16)Wtiny[k * 65 + n] : (_Float16)0.0f;
    }
    for (int i = tid; i < 80; i += 256) sBt[i] = (i < 65) ? btiny[i] : 0.0f;
    __syncthreads();

    const int lane = tid & 31;
    const int wave = tid >> 5;
    const int l15  = lane & 15;
    const int hi   = lane >> 4;     // half-wave: 0 or 1
    const int klo  = hi * 8;        // A-operand k offset (halves)
    const int kb   = hi * 8;        // B-operand k offset (halves)
    _Float16* actw = sAct[wave];

    // ---- Hoist ALL loop-invariant B operands (weights) into VGPRs ----
    // 19 x v16h = 152 VGPRs held live across the tile loop; removes per-tile
    // ds_load + s_wait_dscnt from in front of every WMMA.
    v16h B1[4], B2k0[4], B2k1[4], B3k0, B3k1, Bt[5];
#pragma unroll
    for (int nt = 0; nt < 4; ++nt)
        B1[nt] = ldB(sW1t, (nt * 16 + l15) * 32 + kb);
#pragma unroll
    for (int nt = 0; nt < 4; ++nt) {
        B2k0[nt] = ldB(sW2t, (nt * 16 + l15) * 64 + 0  + kb);
        B2k1[nt] = ldB(sW2t, (nt * 16 + l15) * 64 + 32 + kb);
    }
    B3k0 = ldB(sW3t, l15 * 64 + 0  + kb);
    B3k1 = ldB(sW3t, l15 * 64 + 32 + kb);
#pragma unroll
    for (int nt = 0; nt < 5; ++nt)
        Bt[nt] = ldB(sWtt, (nt * 16 + l15) * 32 + kb);

    // Loop-invariant per-lane output bias (column index fixed across tiles)
    float bias[5];
#pragma unroll
    for (int nt = 0; nt < 5; ++nt) {
        int col = nt * 16 + l15;
        bias[nt] = (col <= 64) ? sBt[col] : 0.0f;
    }

    const float PIF = 3.14159265358979323846f;

    int waveGlobal = blockIdx.x * 8 + wave;
    int waveStride = gridDim.x * 8;

    for (int t = waveGlobal; t < numTiles; t += waveStride) {
        const int pBase = t * 16;
        int p  = pBase + l15;
        int pc = (p < Npts) ? p : (Npts - 1);
        const float* ip = inp + (size_t)pc * 3;
        float x = ip[0], y = ip[1], z = ip[2];

        // ---------------- prior net: positional encoding (12 dims) ----------------
        float e0  = __sinf(x * PIF),        e1  = __sinf(x * 2.0f * PIF),
              e2  = __sinf(x * 4.0f * PIF), e3  = __cosf(x * PIF),
              e4  = __cosf(x * 2.0f * PIF), e5  = __cosf(x * 4.0f * PIF),
              e6  = __sinf(y * PIF),        e7  = __sinf(y * 2.0f * PIF),
              e8  = __sinf(y * 4.0f * PIF), e9  = __cosf(y * PIF),
              e10 = __cosf(y * 2.0f * PIF), e11 = __cosf(y * 4.0f * PIF);

        v16h aE;
#pragma unroll
        for (int i = 0; i < 16; ++i) aE[i] = (_Float16)0.0f;
        if (hi == 0) {   // k = 0..7
            aE[0] = (_Float16)e0; aE[1] = (_Float16)e1; aE[2] = (_Float16)e2;
            aE[3] = (_Float16)e3; aE[4] = (_Float16)e4; aE[5] = (_Float16)e5;
            aE[6] = (_Float16)e6; aE[7] = (_Float16)e7;
        } else {         // k = 8..15 (only 8..11 nonzero)
            aE[0] = (_Float16)e8; aE[1] = (_Float16)e9;
            aE[2] = (_Float16)e10; aE[3] = (_Float16)e11;
        }

        // ---- layer 1: [16,32(pad12)] @ [32,64] ----
        v8f h1a[4];
#pragma unroll
        for (int nt = 0; nt < 4; ++nt)
            h1a[nt] = wmma16(aE, B1[nt], v8f_zero());
        // sigmoid -> LDS act tile [16 rows][64 cols] f16
#pragma unroll
        for (int nt = 0; nt < 4; ++nt)
#pragma unroll
            for (int r = 0; r < 8; ++r)
                actw[(hi * 8 + r) * 64 + nt * 16 + l15] = (_Float16)sigm(h1a[nt][r]);

        // ---- layer 2: [16,64] @ [64,64] ----
        v16h aH0 = ldA(actw, l15 * 64 + 0  + klo);
        v16h aH1 = ldA(actw, l15 * 64 + 32 + klo);
        v8f h2a[4];
#pragma unroll
        for (int nt = 0; nt < 4; ++nt) {
            v8f acc = wmma16(aH0, B2k0[nt], v8f_zero());
            h2a[nt] = wmma16(aH1, B2k1[nt], acc);
        }
#pragma unroll
        for (int nt = 0; nt < 4; ++nt)
#pragma unroll
            for (int r = 0; r < 8; ++r)
                actw[(hi * 8 + r) * 64 + nt * 16 + l15] = (_Float16)sigm(h2a[nt][r]);

        // ---- layer 3: [16,64] @ [64,1(pad16)] -> prior_z in column 0 ----
        v16h aG0 = ldA(actw, l15 * 64 + 0  + klo);
        v16h aG1 = ldA(actw, l15 * 64 + 32 + klo);
        v8f pz = wmma16(aG0, B3k0, v8f_zero());
        pz     = wmma16(aG1, B3k1, pz);

        // ---------------- hash encoding: 8 levels per half-wave ----------------
        float hx = x * (1.0f / 30.0f) + 0.5f;
        float hy = y * (1.0f / 30.0f) + 0.5f;
        v16h fF;
#pragma unroll
        for (int li = 0; li < 8; ++li) {
            int l = hi * 8 + li;
            float s = kScale[l];
            int res = kRes[l];
            float px = hx * s + 0.5f, py = hy * s + 0.5f;
            float fpx = floorf(px), fpy = floorf(py);
            float wx = px - fpx, wy = py - fpy;
            int ix = (int)fpx, iy = (int)fpy;
            int i00, i10, i01, i11;
            if (res <= 724) {  // dense level
                i00 = ix + iy * res; i10 = i00 + 1;
                i01 = i00 + res;     i11 = i01 + 1;
            } else {           // hashed level
                unsigned ux = (unsigned)ix, uy = (unsigned)iy;
                const unsigned m = TSIZE - 1u;
                unsigned hy0 = uy * 2654435761u, hy1 = (uy + 1u) * 2654435761u;
                i00 = (int)((ux        ^ hy0) & m);
                i10 = (int)(((ux + 1u) ^ hy0) & m);
                i01 = (int)((ux        ^ hy1) & m);
                i11 = (int)(((ux + 1u) ^ hy1) & m);
            }
            const int mx = (int)TSIZE - 1;
            i00 = i00 < 0 ? 0 : (i00 > mx ? mx : i00);
            i10 = i10 < 0 ? 0 : (i10 > mx ? mx : i10);
            i01 = i01 < 0 ? 0 : (i01 > mx ? mx : i01);
            i11 = i11 < 0 ? 0 : (i11 > mx ? mx : i11);
            const float2* tb = (const float2*)(tables + (((size_t)l) << LOG2_T) * 2);
            float2 f00 = tb[i00], f10 = tb[i10], f01 = tb[i01], f11 = tb[i11];
            float omx = 1.0f - wx, omy = 1.0f - wy;
            float r0 = (f00.x * omx + f10.x * wx) * omy + (f01.x * omx + f11.x * wx) * wy;
            float r1 = (f00.y * omx + f10.y * wx) * omy + (f01.y * omx + f11.y * wx) * wy;
            fF[li * 2 + 0] = (_Float16)r0;
            fF[li * 2 + 1] = (_Float16)r1;
        }
        // feat tile [16 rows][32 cols] f16 (reuse act buffer; per-lane K-contiguous)
        *(v16h*)(actw + l15 * 32 + hi * 16) = fF;
        v16h aF = ldA(actw, l15 * 32 + klo);

        // ---- tiny layer: [16,32] @ [32,80(pad65)] ----
        v8f xa[5];
#pragma unroll
        for (int nt = 0; nt < 5; ++nt)
            xa[nt] = wmma16(aF, Bt[nt], v8f_zero());

        // ---------------- output: [z - x0 - prior, x1..x64] ----------------
        float zArr[8];
#pragma unroll
        for (int r = 0; r < 8; ++r) zArr[r] = __shfl(z, (hi << 3) + r, 32);

#pragma unroll
        for (int nt = 0; nt < 5; ++nt) {
            int col = nt * 16 + l15;
            if (col > 64) continue;
#pragma unroll
            for (int r = 0; r < 8; ++r) {
                int p2 = pBase + hi * 8 + r;
                if (p2 < Npts) {
                    float val = xa[nt][r] + bias[nt];
                    if (col == 0) val = zArr[r] - val - pz[r];
                    out[(size_t)p2 * 65 + col] = val;
                }
            }
        }
    }
}

extern "C" void kernel_launch(void* const* d_in, const int* in_sizes, int n_in,
                              void* d_out, int out_size, void* d_ws, size_t ws_size,
                              hipStream_t stream) {
    const float* inp    = (const float*)d_in[0];
    const float* tables = (const float*)d_in[1];
    const float* Wtiny  = (const float*)d_in[2];
    const float* btiny  = (const float*)d_in[3];
    const float* Wp1    = (const float*)d_in[4];
    const float* Wp2    = (const float*)d_in[5];
    const float* Wp3    = (const float*)d_in[6];

    int Npts = in_sizes[0] / 3;
    int numTiles = (Npts + 15) / 16;          // 125000 for N=2M
    int blocks = (numTiles + 7) / 8;          // 8 waves (tiles) per 256-thread block
    if (blocks > 4096) blocks = 4096;         // grid-stride over tiles

    hipLaunchKernelGGL(ngp_fused_kernel, dim3(blocks), dim3(256), 0, stream,
                       inp, tables, Wtiny, btiny, Wp1, Wp2, Wp3,
                       (float*)d_out, Npts, numTiles);
}
